// RLNet_6468220748398
// MI455X (gfx1250) — matrix operand order repaired and checked by reading, not compiled, not run on hardware
//
#include <hip/hip_runtime.h>
#include <hip/hip_bf16.h>

typedef __attribute__((ext_vector_type(16))) __bf16 v16bf;
typedef __attribute__((ext_vector_type(8)))  float  v8f;

#define H   128
#define NA  5000
#define FD  32
#define ED  8
#define AD  6
#define NL  3

// ---------- float <-> order-preserving unsigned key (for atomic max) ----------
__device__ __forceinline__ unsigned fkey(float f) {
    unsigned u = __float_as_uint(f);
    return (u & 0x80000000u) ? ~u : (u | 0x80000000u);
}
__device__ __forceinline__ float funkey(unsigned u) {
    return (u & 0x80000000u) ? __uint_as_float(u & 0x7FFFFFFFu) : __uint_as_float(~u);
}

__device__ __forceinline__ void lds_fence() {
    // same-wave LDS RAW turn-around; also a compiler reordering fence
    asm volatile("s_wait_dscnt 0" ::: "memory");
}

// ---------- WMMA fragment helpers (CDNA5 16x16x32 bf16 layouts, ISA 7.12.2) ----------
// A (16x32): lanes 0-15 row M=lane, elems 0-7 -> K 0..7, elems 8-15 -> K 16..23
//            lanes 16-31 row M=lane-16, elems 0-7 -> K 8..15, elems 8-15 -> K 24..31
__device__ __forceinline__ v16bf load_a(const __bf16* base, int stride, int kt) {
    int l   = threadIdx.x & 31;
    int row = l & 15;
    int kb  = kt * 32 + ((l & 16) ? 8 : 0);
    const __bf16* p = base + row * stride + kb;
    v16bf a;
    *((uint4*)&a)     = *(const uint4*)p;
    *((uint4*)&a + 1) = *(const uint4*)(p + 16);
    return a;
}
// B fragments are pre-packed so each lane does one 32B contiguous load
__device__ __forceinline__ v16bf load_b(const __bf16* packed, int NT, int kt, int nt) {
    int l = threadIdx.x & 31;
    return *(const v16bf*)(packed + (((kt * NT + nt) * 32 + l) << 4));
}
__device__ __forceinline__ v8f wmma_bf16(v16bf a, v16bf b, v8f c) {
    return __builtin_amdgcn_wmma_f32_16x16x32_bf16(false, a, false, b, (short)0, c, false, false);
}

// ---------- pack f32 weight [K][N] row-major into B-fragment order, bf16 ----------
__global__ void k_pack(const float* __restrict__ W, __bf16* __restrict__ out, int K, int N) {
    int idx = blockIdx.x * blockDim.x + threadIdx.x;
    if (idx >= K * N) return;
    int e    = idx & 15;
    int lane = (idx >> 4) & 31;
    int t    = idx >> 9;
    int NT   = N >> 4;
    int kt   = t / NT, nt = t % NT;
    int k    = kt * 32 + ((lane & 16) ? 8 : 0) + (e & 7) + ((e & 8) ? 16 : 0);   // B: lanes 0-15 K 0..7,16..23; 16-31 K 8..15,24..31
    int n    = nt * 16 + (lane & 15);
    out[idx] = (__bf16)W[k * N + n];
}

// ---------- node embed MLP: x = relu(xa@W1+b1)@W2+b2  (tiny, VALU) ----------
__global__ void k_embed(const float* __restrict__ xa,
                        const float* __restrict__ W1, const float* __restrict__ b1,
                        const float* __restrict__ W2, const float* __restrict__ b2,
                        float* __restrict__ x, __bf16* __restrict__ xbf) {
    __shared__ float h1[H];
    int n = blockIdx.x, j = threadIdx.x;
    const float* xr = xa + n * FD;
    float a = b1[j];
    #pragma unroll 8
    for (int k = 0; k < FD; k++) a += xr[k] * W1[k * H + j];
    h1[j] = fmaxf(a, 0.f);
    __syncthreads();
    float o = b2[j];
    #pragma unroll 8
    for (int k = 0; k < H; k++) o += h1[k] * W2[k * H + j];
    x[n * H + j]   = o;
    xbf[n * H + j] = (__bf16)o;
}

// ---------- edge embed: h1 = relu(raw@W1+b1) via VALU (K=8), then WMMA h1@W2+b2 ----------
// 32 edges per wave: two A-tiles share every B fragment (2x B reuse).
// Output staged in LDS, then one vectorized 8KB contiguous store per tile.
__global__ void k_eemb(const float* __restrict__ raw,
                       const float* __restrict__ W1, const float* __restrict__ b1,
                       const __bf16* __restrict__ pW2, const float* __restrict__ b2,
                       __bf16* __restrict__ ea, int E) {
    __shared__ __align__(16) __bf16 h1s[4][32][H];
    int wave = threadIdx.x >> 5, lane = threadIdx.x & 31;
    long tile = (long)blockIdx.x * 4 + wave;
    long base = tile * 32;
    if (base >= E) return;
    __bf16 (*h1)[H] = h1s[wave];
    for (int idx = lane; idx < 32 * H; idx += 32) {
        int r = idx >> 7, j = idx & 127;
        const float* er = raw + (base + r) * ED;
        float a = b1[j];
        #pragma unroll
        for (int k = 0; k < ED; k++) a += er[k] * W1[k * H + j];
        h1[r][j] = (__bf16)fmaxf(a, 0.f);
    }
    lds_fence();
    v16bf a0[4], a1[4];
    #pragma unroll
    for (int kt = 0; kt < 4; kt++) {
        a0[kt] = load_a(&h1[0][0], H, kt);
        a1[kt] = load_a(&h1[16][0], H, kt);
    }
    int ch = lane & 15, mh = (lane >> 4) << 3;
    #pragma unroll
    for (int nt = 0; nt < 8; nt++) {
        float bb = b2[nt * 16 + ch];
        v8f acc0 = {bb, bb, bb, bb, bb, bb, bb, bb};
        v8f acc1 = acc0;
        #pragma unroll
        for (int kt = 0; kt < 4; kt++) {
            v16bf bf = load_b(pW2, 8, kt, nt);
            acc0 = wmma_bf16(a0[kt], bf, acc0);
            acc1 = wmma_bf16(a1[kt], bf, acc1);
        }
        #pragma unroll
        for (int r = 0; r < 8; r++) {               // stage result (h1 is free now)
            h1[r + mh][nt * 16 + ch]      = (__bf16)acc0[r];
            h1[16 + r + mh][nt * 16 + ch] = (__bf16)acc1[r];
        }
    }
    lds_fence();
    {
        uint4* dv = (uint4*)(ea + base * H);        // contiguous 8KB tile
        const uint4* sv = (const uint4*)&h1[0][0];
        #pragma unroll
        for (int i = 0; i < 16; i++) dv[lane + i * 32] = sv[lane + i * 32];
    }
}

// ---------- fused fx MLP + scatter-max: msg = mlp(ea); atomic umax into per-type buf ----------
// 32 edges per wave
__global__ void k_fx(const __bf16* __restrict__ ea, const int* __restrict__ dst,
                     const __bf16* __restrict__ pW1, const float* __restrict__ b1,
                     const __bf16* __restrict__ pW2, const float* __restrict__ b2,
                     unsigned* __restrict__ mbuf, int E) {
    __shared__ __align__(16) __bf16 h1s[4][32][H];
    int wave = threadIdx.x >> 5, lane = threadIdx.x & 31;
    long tile = (long)blockIdx.x * 4 + wave;
    long base = tile * 32;
    if (base >= E) return;
    const __bf16* arow = ea + base * H;
    __builtin_prefetch(arow + 32 * H, 0, 1);          // next tile -> global_prefetch_b8
    v16bf a0[4], a1[4];
    #pragma unroll
    for (int kt = 0; kt < 4; kt++) {
        a0[kt] = load_a(arow, H, kt);
        a1[kt] = load_a(arow + 16 * H, H, kt);
    }
    __bf16 (*h1)[H] = h1s[wave];
    int ch = lane & 15, mh = (lane >> 4) << 3;
    #pragma unroll
    for (int nt = 0; nt < 8; nt++) {
        float bb = b1[nt * 16 + ch];
        v8f acc0 = {bb, bb, bb, bb, bb, bb, bb, bb};
        v8f acc1 = acc0;
        #pragma unroll
        for (int kt = 0; kt < 4; kt++) {
            v16bf bf = load_b(pW1, 8, kt, nt);
            acc0 = wmma_bf16(a0[kt], bf, acc0);
            acc1 = wmma_bf16(a1[kt], bf, acc1);
        }
        #pragma unroll
        for (int r = 0; r < 8; r++) {
            h1[r + mh][nt * 16 + ch]      = (__bf16)fmaxf(acc0[r], 0.f);
            h1[16 + r + mh][nt * 16 + ch] = (__bf16)fmaxf(acc1[r], 0.f);
        }
    }
    lds_fence();
    v16bf c0[4], c1[4];
    #pragma unroll
    for (int kt = 0; kt < 4; kt++) {
        c0[kt] = load_a(&h1[0][0], H, kt);
        c1[kt] = load_a(&h1[16][0], H, kt);
    }
    int drow0[8], drow1[8];
    #pragma unroll
    for (int r = 0; r < 8; r++) {
        drow0[r] = dst[base + r + mh];
        drow1[r] = dst[base + 16 + r + mh];
    }
    #pragma unroll
    for (int nt = 0; nt < 8; nt++) {
        float bb = b2[nt * 16 + ch];
        v8f acc0 = {bb, bb, bb, bb, bb, bb, bb, bb};
        v8f acc1 = acc0;
        #pragma unroll
        for (int kt = 0; kt < 4; kt++) {
            v16bf bf = load_b(pW2, 8, kt, nt);
            acc0 = wmma_bf16(c0[kt], bf, acc0);
            acc1 = wmma_bf16(c1[kt], bf, acc1);
        }
        #pragma unroll
        for (int r = 0; r < 8; r++) {
            atomicMax(&mbuf[(long)drow0[r] * H + nt * 16 + ch], fkey(acc0[r]));
            atomicMax(&mbuf[(long)drow1[r] * H + nt * 16 + ch], fkey(acc1[r]));
        }
    }
}

// ---------- x += max over 3 edge-type scatter-max buffers (empty -> 0, PyG semantics) ----------
__global__ void k_combine(float* __restrict__ x, __bf16* __restrict__ xbf,
                          const unsigned* __restrict__ m0, const unsigned* __restrict__ m1,
                          const unsigned* __restrict__ m2, int n) {
    int i = blockIdx.x * blockDim.x + threadIdx.x;
    if (i >= n) return;
    unsigned a = m0[i], b = m1[i], c = m2[i];
    float v0 = a ? funkey(a) : 0.f;
    float v1 = b ? funkey(b) : 0.f;
    float v2 = c ? funkey(c) : 0.f;
    float nx = x[i] + fmaxf(fmaxf(v0, v1), v2);
    x[i]   = nx;
    xbf[i] = (__bf16)nx;
}

// ---------- edge residual update: ea += mlp(cat(ea, x[dst]))  (K=256 -> 128 -> 128) ----------
// 32 edges per wave, 2 waves per block (LDS: 2*(16KB rows + 8KB h1) = 48KB).
// Residual add folded into LDS staging; one vectorized 8KB contiguous store per tile.
__global__ void k_eupd(__bf16* __restrict__ ea, const int* __restrict__ dst,
                       const __bf16* __restrict__ xbf,
                       const __bf16* __restrict__ pW1, const float* __restrict__ b1,
                       const __bf16* __restrict__ pW2, const float* __restrict__ b2, int E) {
    __shared__ __align__(16) __bf16 rows_s[2][32][2 * H];
    __shared__ __align__(16) __bf16 h1s[2][32][H];
    int wave = threadIdx.x >> 5, lane = threadIdx.x & 31;
    long tile = (long)blockIdx.x * 2 + wave;
    long base = tile * 32;
    if (base >= E) return;
    __bf16 (*rows)[2 * H] = rows_s[wave];
    __bf16 (*h1)[H]       = h1s[wave];
    // stage contiguous 32x128 bf16 edge tile into LDS cols [0,128)
    {
        const uint4* src = (const uint4*)(ea + base * H);
        for (int i = lane; i < 512; i += 32) {
            int r = i >> 4, c = (i & 15) * 8;
            *(uint4*)&rows[r][c] = src[i];
        }
    }
    // gather x[dst] rows into LDS cols [128,256): lane -> (row, half), 2 rows per lane
    {
        int r0 = lane >> 1, hh = lane & 1;
        #pragma unroll
        for (int g = 0; g < 2; g++) {
            int r = r0 + g * 16;
            const uint4* xs = (const uint4*)(xbf + (long)dst[base + r] * H + hh * 64);
            uint4* dd = (uint4*)&rows[r][H + hh * 64];
            #pragma unroll
            for (int i = 0; i < 8; i++) dd[i] = xs[i];
        }
    }
    lds_fence();
    v16bf a0[8], a1[8];
    #pragma unroll
    for (int kt = 0; kt < 8; kt++) {
        a0[kt] = load_a(&rows[0][0], 2 * H, kt);
        a1[kt] = load_a(&rows[16][0], 2 * H, kt);
    }
    int ch = lane & 15, mh = (lane >> 4) << 3;
    #pragma unroll
    for (int nt = 0; nt < 8; nt++) {
        float bb = b1[nt * 16 + ch];
        v8f acc0 = {bb, bb, bb, bb, bb, bb, bb, bb};
        v8f acc1 = acc0;
        #pragma unroll
        for (int kt = 0; kt < 8; kt++) {
            v16bf bf = load_b(pW1, 8, kt, nt);
            acc0 = wmma_bf16(a0[kt], bf, acc0);
            acc1 = wmma_bf16(a1[kt], bf, acc1);
        }
        #pragma unroll
        for (int r = 0; r < 8; r++) {
            h1[r + mh][nt * 16 + ch]      = (__bf16)fmaxf(acc0[r], 0.f);
            h1[16 + r + mh][nt * 16 + ch] = (__bf16)fmaxf(acc1[r], 0.f);
        }
    }
    lds_fence();
    v16bf c0[4], c1[4];
    #pragma unroll
    for (int kt = 0; kt < 4; kt++) {
        c0[kt] = load_a(&h1[0][0], H, kt);
        c1[kt] = load_a(&h1[16][0], H, kt);
    }
    #pragma unroll
    for (int nt = 0; nt < 8; nt++) {
        float bb = b2[nt * 16 + ch];
        v8f acc0 = {bb, bb, bb, bb, bb, bb, bb, bb};
        v8f acc1 = acc0;
        #pragma unroll
        for (int kt = 0; kt < 4; kt++) {
            v16bf bf = load_b(pW2, 8, kt, nt);
            acc0 = wmma_bf16(c0[kt], bf, acc0);
            acc1 = wmma_bf16(c1[kt], bf, acc1);
        }
        #pragma unroll
        for (int r = 0; r < 8; r++) {                 // residual add + stage (h1 free now)
            int m = r + mh, c = nt * 16 + ch;
            h1[m][c]      = (__bf16)((float)rows[m][c] + acc0[r]);
            h1[m + 16][c] = (__bf16)((float)rows[m + 16][c] + acc1[r]);
        }
    }
    lds_fence();
    {
        uint4* dv = (uint4*)(ea + base * H);          // contiguous 8KB tile
        const uint4* sv = (const uint4*)&h1[0][0];
        #pragma unroll
        for (int i = 0; i < 16; i++) dv[lane + i * 32] = sv[lane + i * 32];
    }
}

// ---------- field head: out = relu(cat(x,action)@W1+b1) . W2 + b2 ----------
__global__ void k_field(const float* __restrict__ x, const float* __restrict__ act,
                        const float* __restrict__ W1, const float* __restrict__ b1,
                        const float* __restrict__ W2, const float* __restrict__ b2,
                        float* __restrict__ out) {
    __shared__ float red[H];
    int n = blockIdx.x, j = threadIdx.x;
    const float* xr = x + n * H;
    const float* ar = act + n * AD;
    float a = b1[j];
    #pragma unroll 8
    for (int k = 0; k < H; k++) a += xr[k] * W1[k * H + j];
    #pragma unroll
    for (int k = 0; k < AD; k++) a += ar[k] * W1[(H + k) * H + j];
    red[j] = fmaxf(a, 0.f) * W2[j];
    __syncthreads();
    for (int s = 64; s > 0; s >>= 1) {
        if (j < s) red[j] += red[j + s];
        __syncthreads();
    }
    if (j == 0) out[n] = red[0] + b2[0];
}

extern "C" void kernel_launch(void* const* d_in, const int* in_sizes, int n_in,
                              void* d_out, int out_size, void* d_ws, size_t ws_size,
                              hipStream_t stream) {
    const float* x_agent = (const float*)d_in[0];
    const float* action  = (const float*)d_in[1];
    const float* ea_raw[3] = {(const float*)d_in[2], (const float*)d_in[3], (const float*)d_in[4]};
    const int*   dst[3]    = {(const int*)d_in[5], (const int*)d_in[6], (const int*)d_in[7]};
    const float* embW1 = (const float*)d_in[8],  *embB1 = (const float*)d_in[9];
    const float* embW2 = (const float*)d_in[10], *embB2 = (const float*)d_in[11];
    const float* eembW1 = (const float*)d_in[12], *eembB1 = (const float*)d_in[13];
    const float* eembW2 = (const float*)d_in[14], *eembB2 = (const float*)d_in[15];
    const float* fxW1 = (const float*)d_in[16], *fxB1 = (const float*)d_in[17];
    const float* fxW2 = (const float*)d_in[18], *fxB2 = (const float*)d_in[19];
    const float* emW1 = (const float*)d_in[20], *emB1 = (const float*)d_in[21];
    const float* emW2 = (const float*)d_in[22], *emB2 = (const float*)d_in[23];
    const float* fW1 = (const float*)d_in[24], *fB1 = (const float*)d_in[25];
    const float* fW2 = (const float*)d_in[26], *fB2 = (const float*)d_in[27];
    int E[3] = {in_sizes[5], in_sizes[6], in_sizes[7]};

    // workspace carve-up (256B aligned)
    char* ws = (char*)d_ws;
    size_t off = 0;
    auto take = [&](size_t bytes) -> char* {
        char* p = ws + off;
        off = (off + bytes + 255) & ~(size_t)255;
        return p;
    };
    float*   x    = (float*)take((size_t)NA * H * 4);
    __bf16*  xbf  = (__bf16*)take((size_t)NA * H * 2);
    __bf16*  ea[3];
    for (int t = 0; t < 3; t++) ea[t] = (__bf16*)take((size_t)E[t] * H * 2);
    unsigned* mbuf = (unsigned*)take((size_t)3 * NA * H * 4);
    const int PHH  = H * H;        // 16384 packed elems for HxH
    const int P2HH = 2 * H * H;    // 32768 packed elems for 2HxH
    __bf16* pEemb = (__bf16*)take((size_t)PHH * 2);
    __bf16* pFx1  = (__bf16*)take((size_t)NL * PHH * 2);
    __bf16* pFx2  = (__bf16*)take((size_t)NL * PHH * 2);
    __bf16* pEm1  = (__bf16*)take((size_t)NL * P2HH * 2);
    __bf16* pEm2  = (__bf16*)take((size_t)NL * PHH * 2);

    auto pack = [&](const float* W, __bf16* P, int K, int N) {
        k_pack<<<(K * N + 255) / 256, 256, 0, stream>>>(W, P, K, N);
    };
    pack(eembW2, pEemb, H, H);
    for (int l = 0; l < NL; l++) {
        pack(fxW1 + (size_t)l * PHH,  pFx1 + (size_t)l * PHH,  H,     H);
        pack(fxW2 + (size_t)l * PHH,  pFx2 + (size_t)l * PHH,  H,     H);
        pack(emW1 + (size_t)l * P2HH, pEm1 + (size_t)l * P2HH, 2 * H, H);
        pack(emW2 + (size_t)l * PHH,  pEm2 + (size_t)l * PHH,  H,     H);
    }

    k_embed<<<NA, H, 0, stream>>>(x_agent, embW1, embB1, embW2, embB2, x, xbf);

    for (int t = 0; t < 3; t++) {
        int tiles = (E[t] + 31) / 32;
        k_eemb<<<(tiles + 3) / 4, 128, 0, stream>>>(ea_raw[t], eembW1, eembB1, pEemb, eembB2,
                                                    ea[t], E[t]);
    }

    for (int l = 0; l < NL; l++) {
        hipMemsetAsync(mbuf, 0, (size_t)3 * NA * H * 4, stream);
        for (int t = 0; t < 3; t++) {
            int tiles = (E[t] + 31) / 32;
            k_fx<<<(tiles + 3) / 4, 128, 0, stream>>>(
                ea[t], dst[t],
                pFx1 + (size_t)l * PHH, fxB1 + (size_t)l * H,
                pFx2 + (size_t)l * PHH, fxB2 + (size_t)l * H,
                mbuf + (size_t)t * NA * H, E[t]);
        }
        k_combine<<<(NA * H + 255) / 256, 256, 0, stream>>>(
            x, xbf, mbuf, mbuf + (size_t)NA * H, mbuf + (size_t)2 * NA * H, NA * H);
        if (l < NL - 1) {  // last-layer edge update cannot affect the output
            for (int t = 0; t < 3; t++) {
                int tiles = (E[t] + 31) / 32;
                k_eupd<<<(tiles + 1) / 2, 64, 0, stream>>>(
                    ea[t], dst[t], xbf,
                    pEm1 + (size_t)l * P2HH, emB1 + (size_t)l * H,
                    pEm2 + (size_t)l * PHH,  emB2 + (size_t)l * H, E[t]);
            }
        }
    }

    k_field<<<NA, H, 0, stream>>>(x, action, fW1, fB1, fW2, fB2, (float*)d_out);
}
